// GRU_66529043415280
// MI455X (gfx1250) — matrix operand order repaired and checked by reading, not compiled, not run on hardware
//
#include <hip/hip_runtime.h>
#include <hip/hip_bf16.h>
#include <math.h>

// Problem constants (from reference)
static constexpr int VV   = 10000;
static constexpr int EE   = 200;
static constexpr int HH   = 512;
static constexpr int SS   = 35;
static constexpr int BB   = 128;
static constexpr int EPAD = 224;          // E padded to multiple of 32 (WMMA K step)
static constexpr int SB   = SS * BB;      // 4480 rows (multiple of 16)

typedef __attribute__((ext_vector_type(16))) __bf16       v16bf;
typedef __attribute__((ext_vector_type(8)))  float        v8f;
typedef __attribute__((ext_vector_type(4)))  unsigned int u32x4;

union Frag {
    u32x4 q[2];
    v16bf v;
};

__device__ __forceinline__ unsigned short f2bf(float f) {
    unsigned int u = __float_as_uint(f);
    unsigned int lsb = (u >> 16) & 1u;
    u += 0x7fffu + lsb;                  // round-to-nearest-even
    return (unsigned short)(u >> 16);
}

// ---------------------------------------------------------------------------
// Small-step WMMA GEMM (register-direct, dual-pair, fused activation):
// out[M,N] = act( sum_p A_p[M,K_p] @ B_p^T + bias + addend )
// One 16x16 C tile per wave; used on the sequential recurrence (M=128).
// ---------------------------------------------------------------------------
__global__ __launch_bounds__(256) void wmma_gemm(
    const unsigned short* __restrict__ A1, const unsigned short* __restrict__ B1t, int K1,
    const unsigned short* __restrict__ A2, const unsigned short* __restrict__ B2t, int K2,
    const float* __restrict__ addend,   // [M,N] or null
    const float* __restrict__ bias,     // [N] or null
    float* __restrict__ out, int M, int N, int act)  // act: 0=none 1=sigmoid 2=tanh
{
    const int wave  = threadIdx.x >> 5;          // wave32
    const int lane  = threadIdx.x & 31;
    const int tileN = blockIdx.x * 8 + wave;     // wave-uniform -> EXEC stays all-ones
    const int tileM = blockIdx.y;
    if (tileN * 16 >= N) return;

    const int hi = (lane >> 4) & 1;
    const int lr = lane & 15;

    v8f c = {0.f, 0.f, 0.f, 0.f, 0.f, 0.f, 0.f, 0.f};

    for (int p = 0; p < 2; ++p) {
        const unsigned short* A  = p ? A2  : A1;
        const unsigned short* Bt = p ? B2t : B1t;
        const int K = p ? K2 : K1;
        if (A == nullptr) continue;

        const unsigned short* arow = A  + (size_t)(tileM * 16 + lr) * K + hi * 16;
        const unsigned short* brow = Bt + (size_t)(tileN * 16 + lr) * K + hi * 16;

        for (int k = 0; k < K; k += 32) {
            __builtin_prefetch(arow + k + 256, 0, 0);   // global_prefetch_b8
            __builtin_prefetch(brow + k + 256, 0, 0);
            Frag fa, fb;
            fa.q[0] = *(const u32x4*)(arow + k);        // K k..k+7   (b128)
            fa.q[1] = *(const u32x4*)(arow + k + 8);    // K k+8..k+15
            fb.q[0] = *(const u32x4*)(brow + k);        // K k..k+7
            fb.q[1] = *(const u32x4*)(brow + k + 8);    // K k+8..k+15
            c = __builtin_amdgcn_wmma_f32_16x16x32_bf16(
                    false, fa.v, false, fb.v, (short)0, c, false, false);
        }
    }

    const int n  = tileN * 16 + lr;
    const int mb = tileM * 16 + hi * 8;
    const float bv = bias ? bias[n] : 0.0f;
#pragma unroll
    for (int r = 0; r < 8; ++r) {
        size_t idx = (size_t)(mb + r) * N + n;
        float v = c[r] + bv;
        if (addend) v += addend[idx];
        if (act == 1)      v = 1.0f / (1.0f + __expf(-v));
        else if (act == 2) v = tanhf(v);
        out[idx] = v;
    }
}

// ---------------------------------------------------------------------------
// Big batched WMMA GEMM (out = A @ Bt^T + bias), single pair, K <= 512:
//  - block stages the shared 16xK A tile into LDS ONCE via gfx1250 async copy
//    (global_load_async_to_lds_b128 + s_wait_asynccnt), then all 8 waves read
//    A fragments from LDS (ds_load_b128).
//  - each wave computes 4 consecutive 16x16 N tiles, reusing one A fragment
//    across 4 WMMAs per K-step (4x arithmetic intensity on the B stream).
// Used for the x-side preactivations (K=224) and the output projection (K=512).
// ---------------------------------------------------------------------------
__global__ __launch_bounds__(256) void wmma_gemm_lds(
    const unsigned short* __restrict__ A, const unsigned short* __restrict__ Bt, int K,
    const float* __restrict__ bias, float* __restrict__ out, int M, int N)
{
    __shared__ unsigned short Atile[16 * 512];   // 16KB, K <= 512

    const int tid   = threadIdx.x;
    const int wave  = tid >> 5;
    const int lane  = tid & 31;
    const int tileM = blockIdx.y;

    // ---- async-stage A tile [16 x K] into LDS (cooperative, 16B chunks) ----
    {
        const unsigned short* abase = A + (size_t)(tileM * 16) * K;
        const int chunks = (16 * K) >> 3;        // number of 16-byte chunks
        for (int cch = tid; cch < chunks; cch += 256) {
            const unsigned short* g = abase + cch * 8;
            unsigned ldsa = (unsigned)(size_t)(&Atile[cch * 8]);
            asm volatile("global_load_async_to_lds_b128 %0, %1, off"
                         :: "v"(ldsa), "v"(g) : "memory");
        }
        asm volatile("s_wait_asynccnt 0x0" ::: "memory");
    }
    __syncthreads();

    const int hi = (lane >> 4) & 1;
    const int lr = lane & 15;
    const int Ntiles = N >> 4;
    const int tbase  = (blockIdx.x * 8 + wave) * 4;

    v8f c[4];
#pragma unroll
    for (int s = 0; s < 4; ++s) c[s] = (v8f){0.f,0.f,0.f,0.f,0.f,0.f,0.f,0.f};

    const unsigned short* brow[4];
#pragma unroll
    for (int s = 0; s < 4; ++s) {
        int t  = tbase + s;
        int tc = (t < Ntiles) ? t : (Ntiles - 1);           // clamp for safe loads
        brow[s] = Bt + (size_t)(tc * 16 + lr) * K + hi * 16;
    }
    const unsigned short* arow = &Atile[lr * K + hi * 16];  // LDS pointer

    for (int k = 0; k < K; k += 32) {
        Frag fa;
        fa.q[0] = *(const u32x4*)(arow + k);                // ds_load_b128
        fa.q[1] = *(const u32x4*)(arow + k + 8);
#pragma unroll
        for (int s = 0; s < 4; ++s) {
            __builtin_prefetch(brow[s] + k + 256, 0, 0);
            Frag fb;
            fb.q[0] = *(const u32x4*)(brow[s] + k);
            fb.q[1] = *(const u32x4*)(brow[s] + k + 8);
            c[s] = __builtin_amdgcn_wmma_f32_16x16x32_bf16(
                       false, fa.v, false, fb.v, (short)0, c[s], false, false);
        }
    }

    const int mb = tileM * 16 + hi * 8;
#pragma unroll
    for (int s = 0; s < 4; ++s) {
        int t = tbase + s;
        if (t >= Ntiles) continue;                          // wave-uniform guard
        int n = t * 16 + lr;
        float bv = bias ? bias[n] : 0.0f;
#pragma unroll
        for (int r = 0; r < 8; ++r)
            out[(size_t)(mb + r) * N + n] = c[s][r] + bv;
    }
}

// ---------------------------------------------------------------------------
// Prep / elementwise kernels
// ---------------------------------------------------------------------------

// dst[n*Kpad + k] = bf16(src[(rowOff+k)*srcN + n]), zero-pad k >= Ksrc
__global__ void pack_wt(const float* __restrict__ src, int srcN, int rowOff,
                        int Ksrc, int Kpad, unsigned short* __restrict__ dst, int N)
{
    int i = blockIdx.x * blockDim.x + threadIdx.x;
    if (i >= N * Kpad) return;
    int n = i / Kpad;
    int k = i - n * Kpad;
    float v = (k < Ksrc) ? src[(size_t)(rowOff + k) * srcN + n] : 0.0f;
    dst[(size_t)n * Kpad + k] = f2bf(v);
}

__global__ void gather_embed(const int* __restrict__ inputs, const float* __restrict__ emb,
                             unsigned short* __restrict__ X, int rows, int E, int Epad)
{
    int i = blockIdx.x * blockDim.x + threadIdx.x;
    if (i >= rows * Epad) return;
    int row = i / Epad;
    int e   = i - row * Epad;
    int tok = inputs[row];
    float v = (e < E) ? emb[(size_t)tok * E + e] : 0.0f;
    X[i] = f2bf(v);
}

__global__ void pack_bf(const float* __restrict__ src, unsigned short* __restrict__ dst, int n)
{
    int i = blockIdx.x * blockDim.x + threadIdx.x;
    if (i < n) dst[i] = f2bf(src[i]);
}

__global__ void mul_rh(const float* __restrict__ r, const float* __restrict__ hinit,
                       unsigned short* __restrict__ rh, int n)
{
    int i = blockIdx.x * blockDim.x + threadIdx.x;
    if (i < n) rh[i] = f2bf(r[i] * hinit[i]);
}

// h = (1-z)*h_init + z*hp ; write fp32 state, bf16 state, optional sequence slot
__global__ void gate_combine(const float* __restrict__ z, const float* __restrict__ hp,
                             const float* __restrict__ hinit,
                             float* __restrict__ hf, unsigned short* __restrict__ hbf,
                             unsigned short* __restrict__ hall, int n)
{
    int i = blockIdx.x * blockDim.x + threadIdx.x;
    if (i >= n) return;
    float zi = z[i];
    float h  = (1.0f - zi) * hinit[i] + zi * hp[i];
    hf[i] = h;
    unsigned short b = f2bf(h);
    hbf[i] = b;
    if (hall) hall[i] = b;
}

__global__ void copy_f32(const float* __restrict__ src, float* __restrict__ dst, int n)
{
    int i = blockIdx.x * blockDim.x + threadIdx.x;
    if (i < n) dst[i] = src[i];
}

// ---------------------------------------------------------------------------
extern "C" void kernel_launch(void* const* d_in, const int* in_sizes, int n_in,
                              void* d_out, int out_size, void* d_ws, size_t ws_size,
                              hipStream_t stream)
{
    const int*   inputs = (const int*)  d_in[0];
    const float* hidden = (const float*)d_in[1];   // [2,B,H] fp32 (h_init, constant all steps)
    const float* emb    = (const float*)d_in[2];
    const float* wr0 = (const float*)d_in[3];  const float* br0 = (const float*)d_in[4];
    const float* wz0 = (const float*)d_in[5];  const float* bz0 = (const float*)d_in[6];
    const float* wh0 = (const float*)d_in[7];  const float* bh0 = (const float*)d_in[8];
    const float* wr1 = (const float*)d_in[9];  const float* br1 = (const float*)d_in[10];
    const float* wz1 = (const float*)d_in[11]; const float* bz1 = (const float*)d_in[12];
    const float* wh1 = (const float*)d_in[13]; const float* bh1 = (const float*)d_in[14];
    const float* wout = (const float*)d_in[15]; const float* bout = (const float*)d_in[16];
    float* out = (float*)d_out;

    // ---- workspace carve-up ----
    char* ws = (char*)d_ws;
    size_t off = 0;
    auto alloc = [&](size_t bytes) -> char* {
        off = (off + 255) & ~(size_t)255;
        char* p = ws + off;
        off += bytes;
        return p;
    };
    unsigned short* Xbf    = (unsigned short*)alloc((size_t)SB * EPAD * 2);
    unsigned short* wr0x_t = (unsigned short*)alloc((size_t)HH * EPAD * 2);
    unsigned short* wz0x_t = (unsigned short*)alloc((size_t)HH * EPAD * 2);
    unsigned short* wh0x_t = (unsigned short*)alloc((size_t)HH * EPAD * 2);
    unsigned short* wr0h_t = (unsigned short*)alloc((size_t)HH * HH * 2);
    unsigned short* wz0h_t = (unsigned short*)alloc((size_t)HH * HH * 2);
    unsigned short* wh0h_t = (unsigned short*)alloc((size_t)HH * HH * 2);
    unsigned short* wr1x_t = (unsigned short*)alloc((size_t)HH * HH * 2);
    unsigned short* wz1x_t = (unsigned short*)alloc((size_t)HH * HH * 2);
    unsigned short* wh1x_t = (unsigned short*)alloc((size_t)HH * HH * 2);
    unsigned short* wr1h_t = (unsigned short*)alloc((size_t)HH * HH * 2);
    unsigned short* wz1h_t = (unsigned short*)alloc((size_t)HH * HH * 2);
    unsigned short* wh1h_t = (unsigned short*)alloc((size_t)HH * HH * 2);
    unsigned short* wout_t = (unsigned short*)alloc((size_t)VV * HH * 2);
    float* Gxr = (float*)alloc((size_t)SB * HH * 4);   // x-side gate preacts (+bias folded)
    float* Gxz = (float*)alloc((size_t)SB * HH * 4);
    float* Gxh = (float*)alloc((size_t)SB * HH * 4);
    float* r_f  = (float*)alloc((size_t)BB * HH * 4);
    float* z_f  = (float*)alloc((size_t)BB * HH * 4);
    float* hp_f = (float*)alloc((size_t)BB * HH * 4);
    unsigned short* rh_bf = (unsigned short*)alloc((size_t)BB * HH * 2);
    unsigned short* h_bf  = (unsigned short*)alloc((size_t)2 * BB * HH * 2); // h0 | h1
    float*          h_f   = (float*)alloc((size_t)2 * BB * HH * 4);          // h0 | h1
    unsigned short* h1all = (unsigned short*)alloc((size_t)SB * HH * 2);

    unsigned short* h0_bf = h_bf;
    unsigned short* h1_bf = h_bf + (size_t)BB * HH;
    float* h0_f = h_f;
    float* h1_f = h_f + (size_t)BB * HH;
    const float* h0_init = hidden;
    const float* h1_init = hidden + (size_t)BB * HH;

    const int BH = BB * HH;                              // 65536

    auto gemm = [&](const unsigned short* A1, const unsigned short* B1t, int K1,
                    const unsigned short* A2, const unsigned short* B2t, int K2,
                    const float* addend, const float* bias,
                    float* o, int M, int N, int act) {
        dim3 grid((N / 16 + 7) / 8, M / 16, 1);
        wmma_gemm<<<grid, 256, 0, stream>>>(A1, B1t, K1, A2, B2t, K2, addend, bias, o, M, N, act);
    };
    auto gemm_big = [&](const unsigned short* A, const unsigned short* Bt, int K,
                        const float* bias, float* o, int M, int N) {
        dim3 grid((N / 16 + 31) / 32, M / 16, 1);        // 8 waves x 4 tiles = 32 N-tiles/block
        wmma_gemm_lds<<<grid, 256, 0, stream>>>(A, Bt, K, bias, o, M, N);
    };
    auto pack = [&](const float* src, int srcN, int rowOff, int Ksrc, int Kpad,
                    unsigned short* dst, int N) {
        int total = N * Kpad;
        pack_wt<<<(total + 255) / 256, 256, 0, stream>>>(src, srcN, rowOff, Ksrc, Kpad, dst, N);
    };

    // ---- 1) pack weights (transposed, bf16, K-padded) ----
    pack(wr0, HH, 0,  EE, EPAD, wr0x_t, HH);  pack(wr0, HH, EE, HH, HH, wr0h_t, HH);
    pack(wz0, HH, 0,  EE, EPAD, wz0x_t, HH);  pack(wz0, HH, EE, HH, HH, wz0h_t, HH);
    pack(wh0, HH, 0,  EE, EPAD, wh0x_t, HH);  pack(wh0, HH, EE, HH, HH, wh0h_t, HH);
    pack(wr1, HH, 0,  HH, HH,  wr1x_t, HH);   pack(wr1, HH, HH, HH, HH, wr1h_t, HH);
    pack(wz1, HH, 0,  HH, HH,  wz1x_t, HH);   pack(wz1, HH, HH, HH, HH, wz1h_t, HH);
    pack(wh1, HH, 0,  HH, HH,  wh1x_t, HH);   pack(wh1, HH, HH, HH, HH, wh1h_t, HH);
    pack(wout, VV, 0, HH, HH,  wout_t, VV);

    // ---- 2) init hidden state buffers (bf16 copy of input hidden) ----
    pack_bf<<<(2 * BH + 255) / 256, 256, 0, stream>>>(hidden, h_bf, 2 * BH);

    // ---- 3) gather embeddings for all timesteps ----
    gather_embed<<<(SB * EPAD + 255) / 256, 256, 0, stream>>>(inputs, emb, Xbf, SB, EE, EPAD);

    // ---- 4) batched x-side preactivations for layer 0 (biases folded in) ----
    gemm_big(Xbf, wr0x_t, EPAD, br0, Gxr, SB, HH);
    gemm_big(Xbf, wz0x_t, EPAD, bz0, Gxz, SB, HH);
    gemm_big(Xbf, wh0x_t, EPAD, bh0, Gxh, SB, HH);

    // ---- 5) sequential recurrence ----
    const int eb = (BH + 255) / 256;
    for (int t = 0; t < SS; ++t) {
        const float* gr = Gxr + (size_t)t * BH;
        const float* gz = Gxz + (size_t)t * BH;
        const float* gh = Gxh + (size_t)t * BH;
        // layer 0
        gemm(h0_bf, wr0h_t, HH, nullptr, nullptr, 0, gr, nullptr, r_f, BB, HH, 1);
        gemm(h0_bf, wz0h_t, HH, nullptr, nullptr, 0, gz, nullptr, z_f, BB, HH, 1);
        mul_rh<<<eb, 256, 0, stream>>>(r_f, h0_init, rh_bf, BH);
        gemm(rh_bf, wh0h_t, HH, nullptr, nullptr, 0, gh, nullptr, hp_f, BB, HH, 2);
        gate_combine<<<eb, 256, 0, stream>>>(z_f, hp_f, h0_init, h0_f, h0_bf, nullptr, BH);
        // layer 1 (x = new h0)
        gemm(h0_bf, wr1x_t, HH, h1_bf, wr1h_t, HH, nullptr, br1, r_f, BB, HH, 1);
        gemm(h0_bf, wz1x_t, HH, h1_bf, wz1h_t, HH, nullptr, bz1, z_f, BB, HH, 1);
        mul_rh<<<eb, 256, 0, stream>>>(r_f, h1_init, rh_bf, BH);
        gemm(h0_bf, wh1x_t, HH, rh_bf, wh1h_t, HH, nullptr, bh1, hp_f, BB, HH, 2);
        gate_combine<<<eb, 256, 0, stream>>>(z_f, hp_f, h1_init, h1_f, h1_bf,
                                             h1all + (size_t)t * BH, BH);
    }

    // ---- 6) batched output projection: [S*B,512] @ [512,10000] + bout ----
    gemm_big(h1all, wout_t, HH, bout, out, SB, VV);

    // ---- 7) final hidden [L,B,H] appended after outputs ----
    copy_f32<<<(2 * BH + 255) / 256, 256, 0, stream>>>(h_f, out + (size_t)SB * VV, 2 * BH);
}
